// Net_9655086481488
// MI455X (gfx1250) — matrix operand order repaired and checked by reading, not compiled, not run on hardware
//
#include <hip/hip_runtime.h>
#include <hip/hip_bf16.h>

// CDNA5 / gfx1250, wave32. Single-workgroup fused IndRNN encoder/decoder.
// Reference collapses to ~46 MFLOP + 12MB of weight reads (see round-0
// analysis): encoder only needs (t=0, batch=4095); decoder output only uses
// decoder-batch 0 = enc-layer1 @ t=0. One WGP, LDS-staged, the only real
// GEMM (20x1024 @ 1024x1024) runs on v_wmma_f32_16x16x32_bf16.

typedef __attribute__((ext_vector_type(16))) __bf16         v16bf;
typedef __attribute__((ext_vector_type(8)))  float          v8f;
typedef __attribute__((ext_vector_type(4)))  float          v4f;
typedef __attribute__((ext_vector_type(16))) unsigned short v16u;
typedef __attribute__((ext_vector_type(8)))  unsigned short v8u;
typedef __attribute__((ext_vector_type(8)))  unsigned int   v8ui;

#define HID   1024
#define STEPS 20

__device__ __forceinline__ unsigned short f2bf(float f) {
  unsigned int u = __builtin_bit_cast(unsigned int, f);
  u += 0x7FFFu + ((u >> 16) & 1u);          // round-to-nearest-even
  return (unsigned short)(u >> 16);
}

// Pack two f32 -> one dword of two bf16.
//  - If the toolchain has the hardware packed convert, use it (1 VALU op).
//  - Else: round-half-up (v_add per element) + single v_perm_b32 byte-pack
//    ({hi[3],hi[2],lo[3],lo[2]}) = 3 VALU per pair, vs 6 for software RNE.
__device__ __forceinline__ unsigned int pack2bf(float lo, float hi) {
#if __has_builtin(__builtin_amdgcn_cvt_pk_bf16_f32)
  auto r = __builtin_amdgcn_cvt_pk_bf16_f32(lo, hi);
  static_assert(sizeof(r) == 4, "packed bf16 convert should be 32-bit");
  return __builtin_bit_cast(unsigned int, r);
#else
  unsigned int ulo = __builtin_bit_cast(unsigned int, lo) + 0x8000u;
  unsigned int uhi = __builtin_bit_cast(unsigned int, hi) + 0x8000u;
  return __builtin_amdgcn_perm(uhi, ulo, 0x07060302u);  // v_perm_b32
#endif
}

__device__ __forceinline__ float relu_(float x) { return fmaxf(x, 0.0f); }

// y[row] = (relu?) (W[row,:] . vin + bias[row]); 16 waves x 64 rows, lanes
// stride the 1024-long dot (coalesced 128B wave loads), wave32 shfl reduce.
__device__ __forceinline__ void matvec1024(const float* __restrict__ W,
                                           const float* __restrict__ vin,
                                           const float* __restrict__ bias,
                                           float* __restrict__ vout,
                                           int wave, int lane, bool do_relu) {
  for (int r = 0; r < 64; ++r) {
    const int row = wave * 64 + r;
    const float* wr = W + row * HID;
    float s = 0.f;
#pragma unroll 4
    for (int j = lane; j < HID; j += 32) s += wr[j] * vin[j];
#pragma unroll
    for (int off = 16; off > 0; off >>= 1) s += __shfl_xor(s, off, 32);
    if (lane == 0) {
      s += bias[row];
      vout[row] = do_relu ? relu_(s) : s;
    }
  }
}

__global__ __launch_bounds__(512)
void indrnn_fused(const float* __restrict__ x,
                  const float* __restrict__ enc_w0, const float* __restrict__ enc_b0,
                  const float* __restrict__ enc_w1, const float* __restrict__ enc_b1,
                  const float* __restrict__ dec_w0, const float* __restrict__ dec_u0,
                  const float* __restrict__ dec_b0,
                  const float* __restrict__ dec_w1, const float* __restrict__ dec_u1,
                  const float* __restrict__ dec_b1,
                  const float* __restrict__ out_w,  const float* __restrict__ out_b,
                  float* __restrict__ pre1,      // d_ws scratch: 20*1024 f32
                  float* __restrict__ out)       // 20*2 f32
{
  __shared__ float          sv0[HID];            //  4 KB
  __shared__ float          sv1[HID];            //  4 KB
  __shared__ float          spre0[HID];          //  4 KB
  __shared__ unsigned short shg[STEPS * HID];    // 40 KB  (GEMM A, bf16)
  __shared__ float          shacc[2 * STEPS];    // output reduction

  const int  tid    = threadIdx.x;
  const int  wave   = tid >> 5;
  const int  lane   = tid & 31;
  const int  lh     = lane & 15;
  const bool hiHalf = lane >= 16;

  if (tid < 2 * STEPS) shacc[tid] = 0.f;

  // ---- Encoder, t=0 only (h_{-1}=0 so enc_u* vanish) -----------------------
  const float x0 = x[4095 * 2 + 0];
  const float x1 = x[4095 * 2 + 1];
  for (int h = tid; h < HID; h += 512)
    sv0[h] = relu_(enc_w0[h * 2 + 0] * x0 + enc_w0[h * 2 + 1] * x1 + enc_b0[h]);
  __syncthreads();

  matvec1024(enc_w1, sv0, enc_b1, sv1, wave, lane, true);    // v1 = enc layer1 @ t=0
  __syncthreads();
  matvec1024(dec_w0, sv1, dec_b0, spre0, wave, lane, false); // pre0 (step-invariant)
  __syncthreads();

  // ---- Decoder layer 0: elementwise 20-step recurrence -> bf16 A matrix ----
  {
    const int na = tid, nb = tid + 512;
    const float pa = spre0[na], pb = spre0[nb];
    const float ua = dec_u0[na], ub = dec_u0[nb];
    float ga = 0.f, gb = 0.f;
    for (int p = 0; p < STEPS; ++p) {
      ga = relu_(pa + ua * ga);
      gb = relu_(pb + ub * gb);
      shg[p * HID + na] = f2bf(ga);
      shg[p * HID + nb] = f2bf(gb);
    }
  }
  __syncthreads();

  // ---- pre1(20x1024) = G(20x1024) @ dec_w1^T  via v_wmma_f32_16x16x32_bf16 -
  // M padded 20->32 (2 tiles); 16 waves x 4 N-tiles; K swept in 32-chunks,
  // unrolled x2 so loads of one chunk overlap the WMMA chain of the other.
  v8f acc[2][4];
#pragma unroll
  for (int m = 0; m < 2; ++m)
#pragma unroll
    for (int q = 0; q < 4; ++q)
      acc[m][q] = (v8f){0.f, 0.f, 0.f, 0.f, 0.f, 0.f, 0.f, 0.f};

#pragma unroll 2
  for (int k0 = 0; k0 < HID; k0 += 32) {
    // A fragments from LDS (lane<16 -> K 0..15, hi half -> K 16..31, matches B)
    v16bf afrag[2];
    const int kb = k0 + (hiHalf ? 16 : 0);
#pragma unroll
    for (int m = 0; m < 2; ++m) {
      const int row = m * 16 + lh;
      v16u au = (v16u)0;                        // zero-pad rows 20..31
      if (row < STEPS) {
        const unsigned short* p = &shg[row * HID + kb];
        v8u lo = *(const v8u*)(p);
        v8u hi = *(const v8u*)(p + 8);
        au = __builtin_shufflevector(lo, hi, 0, 1, 2, 3, 4, 5, 6, 7,
                                             8, 9, 10, 11, 12, 13, 14, 15);
      }
      afrag[m] = __builtin_bit_cast(v16bf, au);
    }
#pragma unroll
    for (int q = 0; q < 4; ++q) {
      // B fragment: B[k][n] = dec_w1[n][k]; lane owns column n, 16 contiguous k
      const int    n  = (wave * 4 + q) * 16 + lh;
      const float* wp = dec_w1 + n * HID + k0 + (hiHalf ? 16 : 0);
      __builtin_prefetch(wp + 64, 0, 3);        // -> global_prefetch_b8
      v4f f0 = *(const v4f*)(wp + 0);
      v4f f1 = *(const v4f*)(wp + 4);
      v4f f2 = *(const v4f*)(wp + 8);
      v4f f3 = *(const v4f*)(wp + 12);
      v8ui bu;
      bu[0] = pack2bf(f0[0], f0[1]);
      bu[1] = pack2bf(f0[2], f0[3]);
      bu[2] = pack2bf(f1[0], f1[1]);
      bu[3] = pack2bf(f1[2], f1[3]);
      bu[4] = pack2bf(f2[0], f2[1]);
      bu[5] = pack2bf(f2[2], f2[3]);
      bu[6] = pack2bf(f3[0], f3[1]);
      bu[7] = pack2bf(f3[2], f3[3]);
      v16bf bfrag = __builtin_bit_cast(v16bf, bu);
      acc[0][q] = __builtin_amdgcn_wmma_f32_16x16x32_bf16(
          false, afrag[0], false, bfrag, (short)0, acc[0][q], false, false);
      acc[1][q] = __builtin_amdgcn_wmma_f32_16x16x32_bf16(
          false, afrag[1], false, bfrag, (short)0, acc[1][q], false, false);
    }
  }

  // C layout (ISA 7.12.2): VGPR r -> M = r (+8 for hi half). Spill valid rows.
#pragma unroll
  for (int m = 0; m < 2; ++m)
#pragma unroll
    for (int q = 0; q < 4; ++q)
#pragma unroll
      for (int r = 0; r < 8; ++r) {
        const int row = m * 16 + r + (hiHalf ? 8 : 0);
        if (row < STEPS) {
          const int col = (wave * 4 + q) * 16 + lh;
          pre1[row * HID + col] = acc[m][q][r] + dec_b1[col];
        }
      }
  __threadfence();
  __syncthreads();

  // ---- Decoder layer 1 recurrence + 2-wide output projection ---------------
  {
    const int   na = tid, nb = tid + 512;
    const float ua = dec_u1[na], ub = dec_u1[nb];
    const float w0a = out_w[na],       w0b = out_w[nb];
    const float w1a = out_w[HID + na], w1b = out_w[HID + nb];
    float za = 0.f, zb = 0.f;
    for (int p = 0; p < STEPS; ++p) {
      za = relu_(pre1[p * HID + na] + ua * za);
      zb = relu_(pre1[p * HID + nb] + ub * zb);
      float c0 = w0a * za + w0b * zb;
      float c1 = w1a * za + w1b * zb;
#pragma unroll
      for (int off = 16; off > 0; off >>= 1) {
        c0 += __shfl_xor(c0, off, 32);
        c1 += __shfl_xor(c1, off, 32);
      }
      if (lane == 0) {
        atomicAdd(&shacc[p * 2 + 0], c0);       // ds_add_f32
        atomicAdd(&shacc[p * 2 + 1], c1);
      }
    }
  }
  __syncthreads();
  if (tid < 2 * STEPS) out[tid] = shacc[tid] + out_b[tid & 1];
}

extern "C" void kernel_launch(void* const* d_in, const int* in_sizes, int n_in,
                              void* d_out, int out_size, void* d_ws, size_t ws_size,
                              hipStream_t stream) {
  (void)in_sizes; (void)n_in; (void)out_size; (void)ws_size;
  const float* x      = (const float*)d_in[0];
  const float* enc_w0 = (const float*)d_in[1];
  /* d_in[2] enc_u0: provably unused (t=0 only, h_{-1}=0) */
  const float* enc_b0 = (const float*)d_in[3];
  const float* enc_w1 = (const float*)d_in[4];
  /* d_in[5] enc_u1: unused */
  const float* enc_b1 = (const float*)d_in[6];
  const float* dec_w0 = (const float*)d_in[7];
  const float* dec_u0 = (const float*)d_in[8];
  const float* dec_b0 = (const float*)d_in[9];
  const float* dec_w1 = (const float*)d_in[10];
  const float* dec_u1 = (const float*)d_in[11];
  const float* dec_b1 = (const float*)d_in[12];
  const float* out_w  = (const float*)d_in[13];
  const float* out_b  = (const float*)d_in[14];

  indrnn_fused<<<1, 512, 0, stream>>>(x, enc_w0, enc_b0, enc_w1, enc_b1,
                                      dec_w0, dec_u0, dec_b0,
                                      dec_w1, dec_u1, dec_b1,
                                      out_w, out_b,
                                      (float*)d_ws, (float*)d_out);
}
